// Gemma4Base_78400333021372
// MI455X (gfx1250) — matrix operand order repaired and checked by reading, not compiled, not run on hardware
//
#include <hip/hip_runtime.h>

// ---------------------------------------------------------------------------
// Gemma-style attention block for MI455X (gfx1250, wave32, WMMA + TDM).
// Pipeline: QKV GEMMs (TDM-staged bf16 WMMA) -> RMSNorm+RoPE -> sliding-window
// flash attention (bf16 WMMA QK^T and PV) -> output GEMM.
// ---------------------------------------------------------------------------

typedef __attribute__((ext_vector_type(16))) __bf16       bf16x16;
typedef __attribute__((ext_vector_type(8)))  __bf16       bf16x8;
typedef __attribute__((ext_vector_type(8)))  float        f32x8;
typedef __attribute__((ext_vector_type(4)))  unsigned int u32x4;
typedef __attribute__((ext_vector_type(8)))  unsigned int u32x8;

#define CB   2
#define CS   2048
#define CH   2560
#define CNH  8
#define CNKV 4
#define CD   256
#define CSW  1024

static __device__ inline bf16x16 cat8(bf16x8 lo, bf16x8 hi) {
    bf16x16 r;
#pragma unroll
    for (int i = 0; i < 8; ++i) { r[i] = lo[i]; r[i + 8] = hi[i]; }
    return r;
}

static __device__ inline f32x8 wmma_bf16(bf16x16 a, bf16x16 b, f32x8 c) {
    // D = A(16x32 bf16) x B(32x16 bf16) + C(16x16 f32)
    return __builtin_amdgcn_wmma_f32_16x16x32_bf16(
        false, a, false, b, (short)0, c, false, false);
}

// fp32 LDS -> bf16 A-fragment (two 8-element K groups; call site makes k = 16*half + i, contiguous)
static __device__ inline bf16x16 cvt8x2(const float* plo, const float* phi) {
    bf16x16 r;
#pragma unroll
    for (int i = 0; i < 8; ++i) {
        r[i]     = (__bf16)plo[i];
        r[i + 8] = (__bf16)phi[i];
    }
    return r;
}

// fp32 LDS -> bf16 B-fragment (16 contiguous K values per lane)
static __device__ inline bf16x16 cvt16(const float* p) {
    bf16x16 r;
#pragma unroll
    for (int i = 0; i < 16; ++i) r[i] = (__bf16)p[i];
    return r;
}

// ---------------------------------------------------------------------------
// Tensor Data Mover: async-load a 2D fp32 tile (tile_w x tile_h elements,
// row stride row_stride elements) from global into LDS at lds_off.
// D# built per CDNA5 ISA (groups 0/1; groups 2/3 zeroed,
// higher dims unused since tile_dim2 = 0). data_size=4B. LDS padding: 4 DWORDs
// after every 32 DWORDs -> LDS row stride of 36 floats (bank-conflict-free
// fragment reads). Tracked by TENSORcnt.
// ---------------------------------------------------------------------------
static __device__ inline void tdm_load_2d(unsigned int lds_off, const void* gp,
                                          unsigned int tile_w, unsigned int tile_h,
                                          unsigned long long row_stride) {
    const unsigned long long ga = (unsigned long long)(uintptr_t)gp;
    // ---- group 0: count=1 | lds_addr<<32 ; global_addr | type=2 ----
    const unsigned long long g0lo = 1ull | ((unsigned long long)lds_off << 32);
    const unsigned long long g0hi = (ga & ((1ull << 57) - 1ull)) | (2ull << 62);
    u32x4 g0 = { (unsigned)g0lo, (unsigned)(g0lo >> 32),
                 (unsigned)g0hi, (unsigned)(g0hi >> 32) };
    // ---- group 1 ----
    const unsigned long long td0 = row_stride;   // tensor_dim0 (tile always in-bounds)
    const unsigned long long td1 = 0x100000ull;  // tensor_dim1 (tile always in-bounds)
    const unsigned long long q0 = (2ull << 16)                 // data_size = 4B
                                | (1ull << 20)                 // pad_enable
                                | (4ull << 22)                 // pad_interval: 32 DW
                                | (3ull << 25)                 // pad_amount: 4 DW
                                | ((td0 & 0xFFFFull) << 48);   // tensor_dim0[15:0]
    const unsigned long long q1 = ((td0 >> 16) & 0xFFFFull)    // tensor_dim0[31:16]
                                | ((td1 & 0xFFFFFFFFull) << 16)// tensor_dim1
                                | ((unsigned long long)(tile_w & 0xFFFFu) << 48);
    const unsigned long long q2 = (unsigned long long)(tile_h & 0xFFFFu) // tile_dim1
                                | ((row_stride & 0xFFFFFFFFull) << 32);  // stride0[31:0]
    const unsigned long long q3 = (row_stride >> 32) & 0xFFFFull;        // stride0[47:32]
    u32x8 g1 = { (unsigned)q0, (unsigned)(q0 >> 32), (unsigned)q1, (unsigned)(q1 >> 32),
                 (unsigned)q2, (unsigned)(q2 >> 32), (unsigned)q3, (unsigned)(q3 >> 32) };
    u32x4 gz = { 0u, 0u, 0u, 0u };
    asm volatile("tensor_load_to_lds %0, %1, %2, %3"
                 :: "s"(g0), "s"(g1), "s"(gz), "s"(gz)
                 : "memory");
}

// ---------------------------------------------------------------------------
// GEMM: C[M,N] = A[M,K](f32 row-major) * W[N,K]^T (f32 row-major), f32 out.
// Tiles staged by the Tensor Data Mover (wave 0 issues the DMA descriptors,
// waits on TENSORcnt, block barrier publishes). fp32 LDS -> bf16 fragments.
// Block: 256 threads (8 waves). Block tile 128(M) x 64(N), K step 32.
// Wave grid 4x2, each wave computes a 32x32 tile as 2x2 WMMA fragments.
// Requires: M%128==0, N%64==0, K%32==0 (true for all four uses here).
// ---------------------------------------------------------------------------
__global__ __launch_bounds__(256) void gemm_bf16_tn(
    const float* __restrict__ A, const float* __restrict__ W,
    float* __restrict__ C, int M, int N, int K)
{
    __shared__ float As[128][36];   // 32 data + 4 pad DWORDs per row (TDM pad)
    __shared__ float Bs[64][36];

    const int tid  = threadIdx.x;
    const int lane = tid & 31;
    const int wv   = tid >> 5;
    const int half = lane >> 4;
    const int ln   = lane & 15;
    const int wm   = wv & 3;         // wave M index (0..3)
    const int wn   = wv >> 2;        // wave N index (0..1)
    const int bm   = blockIdx.y * 128;
    const int bn   = blockIdx.x * 64;

    const unsigned int asOff = (unsigned int)(uintptr_t)&As[0][0];
    const unsigned int bsOff = (unsigned int)(uintptr_t)&Bs[0][0];

    f32x8 acc[2][2];
#pragma unroll
    for (int im = 0; im < 2; ++im)
#pragma unroll
        for (int in = 0; in < 2; ++in)
#pragma unroll
            for (int r = 0; r < 8; ++r) acc[im][in][r] = 0.0f;

    for (int k0 = 0; k0 < K; k0 += 32) {
        // ---- TDM-stage both tiles (wave 0 only; descriptors are uniform) ----
        if (wv == 0) {
            tdm_load_2d(asOff, A + (size_t)bm * K + k0, 32u, 128u,
                        (unsigned long long)K);
            tdm_load_2d(bsOff, W + (size_t)bn * K + k0, 32u, 64u,
                        (unsigned long long)K);
            __builtin_amdgcn_s_wait_tensorcnt(0);
        }
        __syncthreads();

        // ---- fragments (fp32 LDS -> bf16) + WMMA ----
        bf16x16 af[2], bfr[2];
#pragma unroll
        for (int im = 0; im < 2; ++im) {
            const int m = wm * 32 + im * 16 + ln;
            af[im] = cvt8x2(&As[m][half * 16], &As[m][half * 16 + 8]);
        }
#pragma unroll
        for (int in = 0; in < 2; ++in) {
            const int n = wn * 32 + in * 16 + ln;
            bfr[in] = cvt16(&Bs[n][half * 16]);
        }
#pragma unroll
        for (int im = 0; im < 2; ++im)
#pragma unroll
            for (int in = 0; in < 2; ++in)
                acc[im][in] = wmma_bf16(af[im], bfr[in], acc[im][in]);
        __syncthreads();   // all waves done reading before next DMA overwrites
    }

    // ---- write C (C-layout: lane = col, VGPR r = row half*8+r) ----
#pragma unroll
    for (int im = 0; im < 2; ++im)
#pragma unroll
        for (int in = 0; in < 2; ++in) {
            const int n = bn + wn * 32 + in * 16 + ln;
#pragma unroll
            for (int r = 0; r < 8; ++r) {
                const int m = bm + wm * 32 + im * 16 + half * 8 + r;
                C[(size_t)m * N + n] = acc[im][in][r];
            }
        }
}

// ---------------------------------------------------------------------------
// RMSNorm (+weight for Q/K, bare for V) + RoPE (Q/K only), fp32 in, bf16 out.
// grid.x = B*S, grid.y = NH + 2*NKV (0..7: Q heads, 8..11: K, 12..15: V).
// Q -> qb[b,h,s,d], K -> kb[b,kv,s,d], V -> vbt[b,kv,d,s] (transposed for AV).
// ---------------------------------------------------------------------------
__global__ __launch_bounds__(256) void norm_rope_kernel(
    const float* __restrict__ q_raw, const float* __restrict__ k_raw,
    const float* __restrict__ v_raw, const float* __restrict__ cosT,
    const float* __restrict__ sinT, const float* __restrict__ qw,
    const float* __restrict__ kw, __bf16* __restrict__ qb,
    __bf16* __restrict__ kb, __bf16* __restrict__ vbt)
{
    __shared__ float red[8];
    __shared__ float ynorm[CD];

    const int bs = blockIdx.x;
    const int b  = bs / CS;
    const int s  = bs % CS;
    const int y  = blockIdx.y;
    const int d  = threadIdx.x;
    const int lane = threadIdx.x & 31;
    const int wv   = threadIdx.x >> 5;

    float x, wf;
    if (y < CNH) {               // Q
        x  = q_raw[(size_t)bs * (CNH * CD) + y * CD + d];
        wf = 1.0f + qw[d];
    } else if (y < CNH + CNKV) { // K
        x  = k_raw[(size_t)bs * (CNKV * CD) + (y - CNH) * CD + d];
        wf = 1.0f + kw[d];
    } else {                     // V
        x  = v_raw[(size_t)bs * (CNKV * CD) + (y - CNH - CNKV) * CD + d];
        wf = 1.0f;
    }

    float ss = x * x;
#pragma unroll
    for (int off = 16; off; off >>= 1) ss += __shfl_xor(ss, off, 32);
    if (lane == 0) red[wv] = ss;
    __syncthreads();
    float tot = 0.0f;
#pragma unroll
    for (int i = 0; i < 8; ++i) tot += red[i];
    const float rstd = rsqrtf(tot * (1.0f / CD) + 1e-6f);

    const float yn = x * rstd * wf;
    ynorm[d] = yn;
    __syncthreads();

    if (y >= CNH + CNKV) {       // V: no rope, transposed store
        const int kvh = y - CNH - CNKV;
        vbt[(((size_t)b * CNKV + kvh) * CD + d) * CS + s] = (__bf16)yn;
        return;
    }
    const float c  = cosT[(size_t)s * CD + d];
    const float sn = sinT[(size_t)s * CD + d];
    const float partner = (d < CD / 2) ? ynorm[d + CD / 2] : ynorm[d - CD / 2];
    const float o = (d < CD / 2) ? (yn * c - partner * sn)
                                 : (yn * c + partner * sn);
    if (y < CNH) {
        qb[(((size_t)b * CNH + y) * CS + s) * CD + d] = (__bf16)o;
    } else {
        kb[(((size_t)b * CNKV + (y - CNH)) * CS + s) * CD + d] = (__bf16)o;
    }
}

// ---------------------------------------------------------------------------
// Sliding-window flash attention, one wave per 16-query tile.
// Block = 4 independent waves (no block-wide barriers: window lengths differ).
// Score tile per iteration: 16 q x 64 k (4 WMMA frags), D=256 -> 8 k-steps.
// Online softmax state is per-row, replicated across each 16-lane half.
// P is transposed C->A layout through a per-wave LDS slab (s_wait_dscnt).
// ---------------------------------------------------------------------------
__global__ __launch_bounds__(128) void attn_kernel(
    const __bf16* __restrict__ qb, const __bf16* __restrict__ kb,
    const __bf16* __restrict__ vbt, float* __restrict__ attn)
{
    __shared__ __bf16 Pl[4][16][72];   // per-wave P staging, 144B row stride

    const int lane = threadIdx.x & 31;
    const int wv   = threadIdx.x >> 5;
    const int half = lane >> 4;
    const int ln   = lane & 15;

    const int gw = blockIdx.x * 4 + wv;          // global wave id
    const int qt = gw % (CS / 16);               // q tile (16 rows)
    const int h  = (gw / (CS / 16)) % CNH;       // q head
    const int b  = gw / ((CS / 16) * CNH);
    const int kv = h >> 1;                       // repeat_interleave grouping

    const __bf16* qbase = qb  + ((size_t)b * CNH  + h)  * CS * CD;
    const __bf16* kbase = kb  + ((size_t)b * CNKV + kv) * CS * CD;
    const __bf16* vbase = vbt + ((size_t)b * CNKV + kv) * CD * (size_t)CS;

    // ---- preload Q fragments (A-layout, 8 chunks of K=32 over D=256) ----
    bf16x16 qa[8];
    {
        const int i = qt * 16 + ln;
#pragma unroll
        for (int c = 0; c < 8; ++c) {
            bf16x8 lo = *(const bf16x8*)(qbase + (size_t)i * CD + c * 32 + half * 16);
            bf16x8 hi = *(const bf16x8*)(qbase + (size_t)i * CD + c * 32 + half * 16 + 8);
            qa[c] = cat8(lo, hi);
        }
    }

    f32x8 O[16];
#pragma unroll
    for (int f = 0; f < 16; ++f)
#pragma unroll
        for (int r = 0; r < 8; ++r) O[f][r] = 0.0f;
    float mrow[8], lrow[8];
#pragma unroll
    for (int r = 0; r < 8; ++r) { mrow[r] = -1e30f; lrow[r] = 0.0f; }

    int jlo = qt * 16 - CSW + 1;
    if (jlo < 0) jlo = 0;
    const int j0_start = (jlo / 64) * 64;
    const int j_end    = qt * 16 + 15;

    for (int j0 = j0_start; j0 <= j_end; j0 += 64) {
        // ---- scores S = Q K^T ----
        f32x8 Sf[4];
#pragma unroll
        for (int n = 0; n < 4; ++n) {
#pragma unroll
            for (int r = 0; r < 8; ++r) Sf[n][r] = 0.0f;
#pragma unroll
            for (int c = 0; c < 8; ++c) {
                bf16x16 bk = *(const bf16x16*)(
                    kbase + (size_t)(j0 + n * 16 + ln) * CD + c * 32 + half * 16);
                Sf[n] = wmma_bf16(qa[c], bk, Sf[n]);
            }
        }

        // ---- mask + online softmax ----
        float mnew[8];
#pragma unroll
        for (int r = 0; r < 8; ++r) {
            const int irow = qt * 16 + half * 8 + r;
            float mx = -1e30f;
#pragma unroll
            for (int n = 0; n < 4; ++n) {
                const int j = j0 + n * 16 + ln;
                const bool ok = (j <= irow) && (j + CSW > irow);
                const float v = ok ? Sf[n][r] : -1e30f;
                Sf[n][r] = v;
                mx = fmaxf(mx, v);
            }
#pragma unroll
            for (int off = 8; off; off >>= 1)
                mx = fmaxf(mx, __shfl_xor(mx, off, 32));
            mnew[r] = fmaxf(mrow[r], mx);
        }
        float alpha[8];
#pragma unroll
        for (int r = 0; r < 8; ++r) {
            alpha[r] = __expf(mrow[r] - mnew[r]);
            mrow[r]  = mnew[r];
        }
#pragma unroll
        for (int n = 0; n < 4; ++n)
#pragma unroll
            for (int r = 0; r < 8; ++r) {
                const float sv = Sf[n][r];
                Sf[n][r] = (sv <= -1e29f) ? 0.0f : __expf(sv - mnew[r]);
            }
#pragma unroll
        for (int r = 0; r < 8; ++r) {
            float ps = Sf[0][r] + Sf[1][r] + Sf[2][r] + Sf[3][r];
#pragma unroll
            for (int off = 8; off; off >>= 1) ps += __shfl_xor(ps, off, 32);
            lrow[r] = lrow[r] * alpha[r] + ps;
        }
#pragma unroll
        for (int f = 0; f < 16; ++f)
#pragma unroll
            for (int r = 0; r < 8; ++r) O[f][r] *= alpha[r];

        // ---- transpose P via per-wave LDS (C-layout -> A-layout, bf16) ----
#pragma unroll
        for (int n = 0; n < 4; ++n)
#pragma unroll
            for (int r = 0; r < 8; ++r)
                Pl[wv][half * 8 + r][n * 16 + ln] = (__bf16)Sf[n][r];
        asm volatile("s_wait_dscnt 0" ::: "memory");   // DS in-order per wave
        __builtin_amdgcn_wave_barrier();

        // ---- O += P V ----
#pragma unroll
        for (int kk = 0; kk < 2; ++kk) {
            bf16x8 plo = *(const bf16x8*)&Pl[wv][ln][kk * 32 + half * 16];
            bf16x8 phi = *(const bf16x8*)&Pl[wv][ln][kk * 32 + half * 16 + 8];
            bf16x16 pa = cat8(plo, phi);
#pragma unroll
            for (int nd = 0; nd < 16; ++nd) {
                bf16x16 bv = *(const bf16x16*)(
                    vbase + (size_t)(nd * 16 + ln) * CS + j0 + kk * 32 + half * 16);
                O[nd] = wmma_bf16(pa, bv, O[nd]);
            }
        }
        asm volatile("" ::: "memory");   // keep next P stores after these loads
    }

    // ---- normalize and write [B*S, NH*D] fp32 for output GEMM ----
    float inv[8];
#pragma unroll
    for (int r = 0; r < 8; ++r) inv[r] = (lrow[r] > 0.0f) ? 1.0f / lrow[r] : 0.0f;
#pragma unroll
    for (int nd = 0; nd < 16; ++nd)
#pragma unroll
        for (int r = 0; r < 8; ++r) {
            const int s = qt * 16 + half * 8 + r;
            attn[((size_t)b * CS + s) * (CNH * CD) + h * CD + nd * 16 + ln] =
                O[nd][r] * inv[r];
        }
}

// ---------------------------------------------------------------------------
// Launch
// ---------------------------------------------------------------------------
extern "C" void kernel_launch(void* const* d_in, const int* in_sizes, int n_in,
                              void* d_out, int out_size, void* d_ws, size_t ws_size,
                              hipStream_t stream) {
    (void)in_sizes; (void)n_in; (void)out_size; (void)ws_size;
    const float* hidden = (const float*)d_in[0];
    // d_in[1] = attention_mask: recomputed analytically (deterministic formula)
    const float* cosT = (const float*)d_in[2];
    const float* sinT = (const float*)d_in[3];
    const float* Wq   = (const float*)d_in[4];
    const float* Wk   = (const float*)d_in[5];
    const float* Wv   = (const float*)d_in[6];
    const float* Wo   = (const float*)d_in[7];
    const float* qw   = (const float*)d_in[8];
    const float* kw   = (const float*)d_in[9];
    float* out = (float*)d_out;

    const size_t M = (size_t)CB * CS;            // 4096
    char* ws = (char*)d_ws;
    float*  q_raw = (float*)ws;   ws += M * (CNH  * CD) * 4;   // 32 MB
    float*  k_raw = (float*)ws;   ws += M * (CNKV * CD) * 4;   // 16 MB
    float*  v_raw = (float*)ws;   ws += M * (CNKV * CD) * 4;   // 16 MB
    __bf16* qb    = (__bf16*)ws;  ws += M * (CNH  * CD) * 2;   // 16 MB
    __bf16* kb    = (__bf16*)ws;  ws += M * (CNKV * CD) * 2;   //  8 MB
    __bf16* vbt   = (__bf16*)ws;  ws += M * (CNKV * CD) * 2;   //  8 MB
    float*  attn  = (float*)ws;                                // 32 MB

    // QKV projections (TDM-staged WMMA GEMMs)
    gemm_bf16_tn<<<dim3((CNH * CD) / 64, 4096 / 128), 256, 0, stream>>>(
        hidden, Wq, q_raw, 4096, CNH * CD, CH);
    gemm_bf16_tn<<<dim3((CNKV * CD) / 64, 4096 / 128), 256, 0, stream>>>(
        hidden, Wk, k_raw, 4096, CNKV * CD, CH);
    gemm_bf16_tn<<<dim3((CNKV * CD) / 64, 4096 / 128), 256, 0, stream>>>(
        hidden, Wv, v_raw, 4096, CNKV * CD, CH);

    // RMSNorm + RoPE, fp32 -> bf16 (V transposed)
    norm_rope_kernel<<<dim3(4096, CNH + 2 * CNKV), 256, 0, stream>>>(
        q_raw, k_raw, v_raw, cosT, sinT, qw, kw, qb, kb, vbt);

    // Sliding-window attention: B*NH*(S/16) = 2048 waves, 4 waves/block
    attn_kernel<<<dim3((CB * CNH * (CS / 16)) / 4), 128, 0, stream>>>(
        qb, kb, vbt, attn);

    // Output projection
    gemm_bf16_tn<<<dim3(CH / 64, 4096 / 128), 256, 0, stream>>>(
        attn, Wo, out, 4096, CH, CNH * CD);
}
